// hybridloss_73358041415728
// MI455X (gfx1250) — compile-verified, matrix operand order checked by reading, not run on hardware
//
#include <hip/hip_runtime.h>
#include <hip/hip_bf16.h>
#include <math.h>

// ---------------- problem constants ----------------
#define BB      512
#define DD      128
#define OUTN    100000
#define S_SCALE 32.0f
#define K2LOG   46.166241308446828384f  // S * log2(e)
#define COS_M   0.8775825618903728f     // cos(0.5)
#define SIN_M   0.479425538604203f      // sin(0.5)
#define TH_C    (-0.8775825618903728f)  // cos(pi-0.5)
#define MM_C    0.2397127693021015f     // sin(pi-0.5)*0.5
#define IMG_N   1024
#define COL_TILES 6250                  // OUTN / 16
#define CT_PER_WAVE 8                   // col tiles per wave (A reuse)
#define CT_PER_BLOCK 64                 // 8 waves * 8

typedef _Float16 v16h   __attribute__((ext_vector_type(16)));
typedef _Float16 half8  __attribute__((ext_vector_type(8)));
typedef _Float16 half4v __attribute__((ext_vector_type(4)));
typedef float    v8f    __attribute__((ext_vector_type(8)));

union ABReg { v16h v; half8 h[2]; };

// raw v_exp_f32 (results below 2^-126 flush; fine for softmax tails)
#define EXP2_RAW(x) __builtin_amdgcn_exp2f(x)

// wave32 reduce via ds_swizzle_b32 (group-of-32: offset = xor_mask<<10 | 0x1f)
__device__ __forceinline__ float wave_reduce_add_f32(float v) {
  v += __int_as_float(__builtin_amdgcn_ds_swizzle(__float_as_int(v), 0x041F)); // SWAPX1
  v += __int_as_float(__builtin_amdgcn_ds_swizzle(__float_as_int(v), 0x081F)); // SWAPX2
  v += __int_as_float(__builtin_amdgcn_ds_swizzle(__float_as_int(v), 0x101F)); // SWAPX4
  v += __int_as_float(__builtin_amdgcn_ds_swizzle(__float_as_int(v), 0x201F)); // SWAPX8
  v += __int_as_float(__builtin_amdgcn_ds_swizzle(__float_as_int(v), 0x401F)); // SWAPX16
  return v;
}

// ---------------- workspace layout (bytes) ----------------
#define WNH_OFF    0u          // 100000*128 f16 = 25,600,000
#define XNH_OFF    25600000u   //    512*128 f16 =    131,072
#define ROWSUM_OFF 25731072u   // 512 f32
#define TCOS_OFF   25733120u   // 512 f32 (cosine at label column)
#define SCAL_OFF   25735168u   // 4 f32 (padded to 256)
#define FFTRE_OFF  25735424u   // 1024*1024 f32
#define FFTIM_OFF  29929728u   // 1024*1024 f32

// =====================================================================
// 1) row-normalize f32 [nrows x 128] -> f16, one wave per row
// =====================================================================
__global__ void rownorm_f16_kernel(const float* __restrict__ src,
                                   _Float16* __restrict__ dst, int nrows) {
  int gwave = (int)((blockIdx.x * blockDim.x + threadIdx.x) >> 5);
  int lane  = (int)(threadIdx.x & 31);
  if (gwave >= nrows) return;                 // wave-uniform
  const float4* p = reinterpret_cast<const float4*>(src + (size_t)gwave * DD);
  float4 v = p[lane];
  float ss = v.x * v.x + v.y * v.y + v.z * v.z + v.w * v.w;
  ss = wave_reduce_add_f32(ss);
  float r = rsqrtf(ss);
  half4v h;
  h[0] = (_Float16)(v.x * r);
  h[1] = (_Float16)(v.y * r);
  h[2] = (_Float16)(v.z * r);
  h[3] = (_Float16)(v.w * r);
  *reinterpret_cast<half4v*>(dst + (size_t)gwave * DD + lane * 4) = h;
}

// =====================================================================
// 1b) label cosines: tcos[b] = dot(xn[b], wn[label[b]]), one wave per row
// =====================================================================
__global__ void tcos_kernel(const _Float16* __restrict__ xnH,
                            const _Float16* __restrict__ wnH,
                            const int* __restrict__ label,
                            float* __restrict__ tcos) {
  int gwave = (int)((blockIdx.x * blockDim.x + threadIdx.x) >> 5);
  int lane  = (int)(threadIdx.x & 31);
  if (gwave >= BB) return;                    // wave-uniform
  int lbl = label[gwave];
  const half4v* xp = reinterpret_cast<const half4v*>(xnH + (size_t)gwave * DD);
  const half4v* wp = reinterpret_cast<const half4v*>(wnH + (size_t)lbl * DD);
  half4v xv = xp[lane];
  half4v wv = wp[lane];
  float d = (float)xv[0] * (float)wv[0] + (float)xv[1] * (float)wv[1]
          + (float)xv[2] * (float)wv[2] + (float)xv[3] * (float)wv[3];
  d = wave_reduce_add_f32(d);
  if (lane == 0) tcos[gwave] = d;
}

// =====================================================================
// 2) Fused ArcMargin GEMM.
//    Per wave: A fragment (16x128 f16) loaded ONCE; stream 8 column
//    tiles with double-buffered B fragments; 4 WMMAs per tile.
//    Per element: exactly v_fmaak + v_exp + v_add (no compares, no
//    label handling). Cross-lane combine: ONE ds_add_f32 pass per wave.
// =====================================================================
__global__ void arc_gemm_kernel(const _Float16* __restrict__ xnH,
                                const _Float16* __restrict__ wnH,
                                float* __restrict__ rowSum) {
  __shared__ float blockSum[16];
  if (threadIdx.x < 16) blockSum[threadIdx.x] = 0.0f;
  __syncthreads();

  int wave    = (int)(threadIdx.x >> 5);
  int lane    = (int)(threadIdx.x & 31);
  int laneLo  = lane & 15;
  int laneHi  = lane >> 4;
  int rowTile = (int)blockIdx.y;
  int colBase = (int)blockIdx.x * CT_PER_BLOCK + wave * CT_PER_WAVE;

  // ---- A fragment: 16 rows of xn, ISA 16-bit A layout, kept in VGPRs ----
  const _Float16* arow = xnH + (size_t)(rowTile * 16 + laneLo) * DD;
  ABReg a[4];
#pragma unroll
  for (int kc = 0; kc < 4; ++kc) {
    const half8* ap = reinterpret_cast<const half8*>(arow + kc * 32 + laneHi * 8);
    a[kc].h[0] = ap[0];            // K0..K0+7
    a[kc].h[1] = ap[2];            // K0+16..K0+23
  }

  // ---- per-lane, per-row softmax partial sums (registers) ----
  float acc[8] = {0.f, 0.f, 0.f, 0.f, 0.f, 0.f, 0.f, 0.f};

  auto loadB = [&](ABReg* b, int colTile) {
    const _Float16* brow = wnH + (size_t)(colTile * 16 + laneLo) * DD;
#pragma unroll
    for (int kc = 0; kc < 4; ++kc) {
      const half8* bp = reinterpret_cast<const half8*>(brow + kc * 32 + laneHi * 8);
      b[kc].h[0] = bp[0];
      b[kc].h[1] = bp[2];
    }
  };

  ABReg b[2][4];                        // ping-pong B fragments
  if (colBase < COL_TILES) loadB(b[0], colBase);

#pragma unroll
  for (int ct = 0; ct < CT_PER_WAVE; ++ct) {
    int colTile = colBase + ct;
    if (colTile >= COL_TILES) break;    // wave-uniform
    if (ct + 1 < CT_PER_WAVE && colTile + 1 < COL_TILES)
      loadB(b[(ct + 1) & 1], colTile + 1);   // prefetch next B

    v8f c = {};
#pragma unroll
    for (int kc = 0; kc < 4; ++kc)
      c = __builtin_amdgcn_wmma_f32_16x16x32_f16(
          false, a[kc].v, false, b[ct & 1][kc].v, (short)0, c, false, false);

#pragma unroll
    for (int j = 0; j < 8; ++j)
      acc[j] += EXP2_RAW(fmaf(K2LOG, c[j], -K2LOG)); // exp(S*cos - S)
  }

  // ---- one LDS-atomic combine per lane, then one global flush per block ----
#pragma unroll
  for (int j = 0; j < 8; ++j)
    atomicAdd(&blockSum[j + 8 * laneHi], acc[j]);
  __syncthreads();
  if (threadIdx.x < 16)
    atomicAdd(&rowSum[rowTile * 16 + threadIdx.x], blockSum[threadIdx.x]);
}

// =====================================================================
// 3) TV + binarization loss partial sums over the 1024x1024 image
// =====================================================================
__global__ void tvbin_kernel(const float* __restrict__ k, float* __restrict__ scal) {
  int   idx    = (int)(blockIdx.x * blockDim.x + threadIdx.x);
  int   stride = (int)(gridDim.x * blockDim.x);
  float sh = 0.f, sw = 0.f, sb = 0.f;
  for (int i = idx; i < IMG_N * IMG_N; i += stride) {
    int   r = i >> 10, cidx = i & 1023;
    float v = k[i];
    sb += v * v + (v - 1.0f) * (v - 1.0f);
    if (r < IMG_N - 1)    { float d = k[i + IMG_N] - v; sh += d * d; }
    if (cidx < IMG_N - 1) { float d = k[i + 1]     - v; sw += d * d; }
  }
  sh = wave_reduce_add_f32(sh);
  sw = wave_reduce_add_f32(sw);
  sb = wave_reduce_add_f32(sb);
  if ((threadIdx.x & 31) == 0) {
    atomicAdd(&scal[0], sh);
    atomicAdd(&scal[1], sw);
    atomicAdd(&scal[2], sb);
  }
}

// =====================================================================
// 4) 1024-pt radix-2 DIT FFT in LDS (shared helper)
// =====================================================================
__device__ __forceinline__ void fft1024_lds(float* re, float* im, int tid) {
  for (int len = 2; len <= IMG_N; len <<= 1) {
    int half = len >> 1;
    for (int t = tid; t < IMG_N / 2; t += 256) {
      int   grp = t / half, pos = t - grp * half;
      int   i0  = grp * len + pos, i1 = i0 + half;
      float ang = -6.2831853071795864769f * (float)pos / (float)len;
      float s, cc;
      __sincosf(ang, &s, &cc);
      float vr = re[i1] * cc - im[i1] * s;
      float vi = re[i1] * s  + im[i1] * cc;
      float ur = re[i0], ui = im[i0];
      re[i0] = ur + vr;  im[i0] = ui + vi;
      re[i1] = ur - vr;  im[i1] = ui - vi;
    }
    __syncthreads();
  }
}

__global__ void fft_rows_kernel(const float* __restrict__ k,
                                float* __restrict__ fre, float* __restrict__ fim) {
  __shared__ float re[IMG_N];
  __shared__ float im[IMG_N];
  int row = (int)blockIdx.x, tid = (int)threadIdx.x;
  for (int i = tid; i < IMG_N; i += 256) {
    int j = (int)(__brev((unsigned)i) >> 22);   // 10-bit reversal
    re[j] = k[(size_t)row * IMG_N + i];
    im[j] = 0.0f;
  }
  __syncthreads();
  fft1024_lds(re, im, tid);
  for (int i = tid; i < IMG_N; i += 256) {
    fre[(size_t)row * IMG_N + i] = re[i];
    fim[(size_t)row * IMG_N + i] = im[i];
  }
}

__global__ void fft_cols_kernel(const float* __restrict__ fre,
                                const float* __restrict__ fim,
                                float* __restrict__ scal) {
  __shared__ float re[IMG_N];
  __shared__ float im[IMG_N];
  int col = (int)blockIdx.x, tid = (int)threadIdx.x;
  for (int i = tid; i < IMG_N; i += 256) {
    int j = (int)(__brev((unsigned)i) >> 22);
    re[j] = fre[(size_t)i * IMG_N + col];
    im[j] = fim[(size_t)i * IMG_N + col];
  }
  __syncthreads();
  fft1024_lds(re, im, tid);
  float s = 0.0f;
  for (int i = tid; i < IMG_N; i += 256)
    s += sqrtf(re[i] * re[i] + im[i] * im[i]);
  s = wave_reduce_add_f32(s);
  if ((threadIdx.x & 31) == 0) atomicAdd(&scal[3], s);
}

// =====================================================================
// 5) finalize: apply ArcFace margin per row (512 rows), fold logsumexp
//    correction, reduce, and combine with tv/bin/fft -> d_out[0]
// =====================================================================
__global__ void finalize_kernel(const float* __restrict__ rowSum,
                                const float* __restrict__ tcos,
                                const float* __restrict__ scal,
                                float* __restrict__ out) {
  __shared__ float sdata[BB];
  int t = (int)threadIdx.x;
  float cosv = tcos[t];
  // plain-logit exp that the GEMM accumulated for the label column
  float e0   = EXP2_RAW(fmaf(K2LOG, cosv, -K2LOG));
  // margin-modified logit
  float sine = sqrtf(fmaxf(1.0f - cosv * cosv, 0.0f));
  float phi  = cosv * COS_M - sine * SIN_M;
  float modv = (cosv - TH_C > 0.0f) ? phi : (cosv - MM_C);
  float lg   = S_SCALE * modv;
  float et   = __expf(lg - S_SCALE);
  // -logp[b,label] = (S + log(sum - e0 + et)) - lg
  sdata[t] = S_SCALE + logf(rowSum[t] - e0 + et) - lg;
  __syncthreads();
  for (int s = BB / 2; s > 0; s >>= 1) {
    if (t < s) sdata[t] += sdata[t + s];
    __syncthreads();
  }
  if (t == 0) {
    float arc  = sdata[0] / (float)BB;
    float cnt  = (float)((IMG_N - 1) * IMG_N);
    float tv   = -2.0f * (scal[0] / cnt + scal[1] / cnt);
    float binl = scal[2] / (float)(IMG_N * IMG_N);
    out[0] = arc + tv + binl + scal[3];
  }
}

// =====================================================================
extern "C" void kernel_launch(void* const* d_in, const int* in_sizes, int n_in,
                              void* d_out, int out_size, void* d_ws, size_t ws_size,
                              hipStream_t stream) {
  const float* x     = (const float*)d_in[0];   // [512,128]
  const int*   label = (const int*)d_in[1];     // [512]
  const float* kimg  = (const float*)d_in[2];   // [1,1,1024,1024]
  const float* W     = (const float*)d_in[5];   // [100000,128]
  float*       out   = (float*)d_out;

  char* ws = (char*)d_ws;
  _Float16* wnH    = (_Float16*)(ws + WNH_OFF);
  _Float16* xnH    = (_Float16*)(ws + XNH_OFF);
  float*    rowSum = (float*)(ws + ROWSUM_OFF);
  float*    tcos   = (float*)(ws + TCOS_OFF);
  float*    scal   = (float*)(ws + SCAL_OFF);
  float*    fre    = (float*)(ws + FFTRE_OFF);
  float*    fim    = (float*)(ws + FFTIM_OFF);

  // zero per-call accumulators (rowSum | tcos | scalars are contiguous)
  hipMemsetAsync(rowSum, 0, 2048 + 2048 + 256, stream);

  // normalize W and x into f16 scratch (one wave per row, 8 waves/block)
  rownorm_f16_kernel<<<(OUTN + 7) / 8, 256, 0, stream>>>(W, wnH, OUTN);
  rownorm_f16_kernel<<<(BB + 7) / 8,   256, 0, stream>>>(x, xnH, BB);

  // label-column cosines (512 short dot products)
  tcos_kernel<<<(BB + 7) / 8, 256, 0, stream>>>(xnH, wnH, label, tcos);

  // fused WMMA GEMM + shifted-softmax accumulation (margin deferred)
  dim3 ggrid((COL_TILES + CT_PER_BLOCK - 1) / CT_PER_BLOCK, BB / 16);
  arc_gemm_kernel<<<ggrid, 256, 0, stream>>>(xnH, wnH, rowSum);

  // tv + bin partial sums
  tvbin_kernel<<<1024, 256, 0, stream>>>(kimg, scal);

  // 2D FFT magnitude sum
  fft_rows_kernel<<<IMG_N, 256, 0, stream>>>(kimg, fre, fim);
  fft_cols_kernel<<<IMG_N, 256, 0, stream>>>(fre, fim, scal);

  // combine
  finalize_kernel<<<1, BB, 0, stream>>>(rowSum, tcos, scal, out);
}
